// spline_85667417686111
// MI455X (gfx1250) — matrix operand order repaired and checked by reading, not compiled, not run on hardware
//
#include <hip/hip_runtime.h>
#include <cstring>
#include <cstddef>

typedef __attribute__((ext_vector_type(16))) _Float16 v16h;
typedef __attribute__((ext_vector_type(8)))  _Float16 v8h;
typedef __attribute__((ext_vector_type(8)))  float    v8f;

static constexpr int NPT = 16384;   // nodes level 0
static constexpr int NE  = 65536;   // edges (constant across levels; emask gates)
static constexpr int BG  = 4;       // graphs in batch

// ---------- helpers: ordered-uint encoding for float atomic max ----------
__device__ __forceinline__ unsigned fenc(float f) {
  unsigned u = __float_as_uint(f);
  return (u & 0x80000000u) ? ~u : (u | 0x80000000u);
}
__device__ __forceinline__ float fdec(unsigned u) {
  return (u & 0x80000000u) ? __uint_as_float(u ^ 0x80000000u) : __uint_as_float(~u);
}

// ---------- elementwise / utility kernels ----------
__global__ void fill_u32_kernel(unsigned* __restrict__ p, unsigned v, size_t n) {
  size_t i = (size_t)blockIdx.x * blockDim.x + threadIdx.x;
  if (i < n) p[i] = v;
}

// Per-edge: degree-1 open B-spline basis (K=3, dim=3) -> 8 corner weights + kernel idx.
// Also accumulates mean-aggregation degree deg[dst] += emask.
__global__ void basis_kernel(const float* __restrict__ u, const float* __restrict__ emask,
                             const int* __restrict__ dst, float* __restrict__ basis,
                             int* __restrict__ kidx, float* __restrict__ deg, int ne) {
  int e = blockIdx.x * blockDim.x + threadIdx.x;
  if (e >= ne) return;
  float em = emask ? emask[e] : 1.0f;
  float lo[3], fr[3];
#pragma unroll
  for (int d = 0; d < 3; ++d) {
    float v = u[(size_t)e * 3 + d] * 2.0f;           // u * (K-1)
    float l = floorf(v);
    l = fminf(fmaxf(l, 0.0f), 1.0f);                 // clip to [0, K-2]
    lo[d] = l;
    fr[d] = v - l;
  }
#pragma unroll
  for (int s = 0; s < 8; ++s) {
    float w = em;
    int ki = 0, p = 1;
#pragma unroll
    for (int d = 0; d < 3; ++d) {
      int bb = (s >> d) & 1;
      w *= bb ? fr[d] : (1.0f - fr[d]);
      ki += ((int)lo[d] + bb) * p;
      p *= 3;
    }
    basis[(size_t)e * 8 + s] = w;
    kidx[(size_t)e * 8 + s] = ki;
  }
  atomicAdd(&deg[dst[e]], em);
}

// Edge scatter: S[(dst*27+k)*ci + c] += basis * X[src*ci + c]   (one thread per edge-channel)
__global__ void scatter_kernel(const float* __restrict__ X, const int* __restrict__ src,
                               const int* __restrict__ dst, const float* __restrict__ basis,
                               const int* __restrict__ kidx, float* __restrict__ S,
                               int ci, size_t total) {
  size_t idx = (size_t)blockIdx.x * blockDim.x + threadIdx.x;
  if (idx >= total) return;
  int e = (int)(idx / ci);
  int c = (int)(idx % ci);
  float xv = X[(size_t)src[e] * ci + c];
  int d = dst[e];
#pragma unroll
  for (int j = 0; j < 8; ++j) {
    float b = basis[(size_t)e * 8 + j];
    if (b != 0.0f) {
      int k = kidx[(size_t)e * 8 + j];
      atomicAdd(&S[((size_t)d * 27 + k) * ci + c], b * xv);
    }
  }
}

// Pack concat(W.reshape(27*ci,co), Wr) into f16, stored column-major: Wp[ncol*Kd + k].
// Rows [k1+ci, Kd) zero-padded so the WMMA K-loop runs in steps of 32.
__global__ void pack_w_kernel(const float* __restrict__ W, const float* __restrict__ Wr,
                              _Float16* __restrict__ Wp, int k1, int ci, int Kd, int co) {
  size_t idx = (size_t)blockIdx.x * blockDim.x + threadIdx.x;
  size_t total = (size_t)co * Kd;
  if (idx >= total) return;
  int ncol = (int)(idx / Kd);
  int k = (int)(idx % Kd);
  float v = 0.0f;
  if (k < k1) v = W[(size_t)k * co + ncol];
  else if (k < k1 + ci) v = Wr[(size_t)(k - k1) * co + ncol];
  Wp[(size_t)ncol * Kd + k] = (_Float16)v;
}

// Pack virtual A into f16, row-major (n16 x Kd):
//   k<k1      -> S[row*k1+k] / max(deg[row],1)   (mean-aggregated spline messages)
//   k<k1+ci   -> X[row*ci + k-k1]                (root-weight input features)
//   else / row>=n -> 0                           (K and M padding)
__global__ void pack_a_kernel(const float* __restrict__ S, const float* __restrict__ X,
                              const float* __restrict__ deg, _Float16* __restrict__ Ap,
                              int n, int k1, int ci, int Kd, size_t total) {
  size_t idx = (size_t)blockIdx.x * blockDim.x + threadIdx.x;
  if (idx >= total) return;
  int row = (int)(idx / Kd);
  int k = (int)(idx % Kd);
  float v = 0.0f;
  if (row < n) {
    if (k < k1)
      v = S[(size_t)row * k1 + k] * (deg ? (1.0f / fmaxf(deg[row], 1.0f)) : 1.0f);
    else if (k < k1 + ci)
      v = X[(size_t)row * ci + (k - k1)];
  }
  Ap[idx] = (_Float16)v;
}

// ---------- WMMA GEMM, 16x64 strip per wave (4 accumulators share one A fragment) ----------
// mode 0: H = BN(ELU(acc + bias));  mode 1: H = ELU(acc + bias).
__global__ void __launch_bounds__(32)
wmma_gemm_kernel(const _Float16* __restrict__ Ap, const _Float16* __restrict__ Wp,
                 const float* __restrict__ bias, const float* __restrict__ bng,
                 const float* __restrict__ bnb, const float* __restrict__ bnm,
                 const float* __restrict__ bnv, float* __restrict__ H,
                 int n, int Kd, int co, int mode) {
  const int lane = threadIdx.x;
  const int lm = lane & 15;
  const int hi = lane >> 4;
  const int arow = blockIdx.y * 16 + lm;       // A fragment: lane -> M row (padded rows are 0)
  const int bcol0 = blockIdx.x * 64 + lm;      // B fragments: 4 N-tiles, 16 apart
  const _Float16* ap = Ap + (size_t)arow * Kd;
  const _Float16* w0 = Wp + (size_t)(bcol0)      * Kd;
  const _Float16* w1 = Wp + (size_t)(bcol0 + 16) * Kd;
  const _Float16* w2 = Wp + (size_t)(bcol0 + 32) * Kd;
  const _Float16* w3 = Wp + (size_t)(bcol0 + 48) * Kd;
  v8f acc0 = {0.f,0.f,0.f,0.f,0.f,0.f,0.f,0.f};
  v8f acc1 = acc0, acc2 = acc0, acc3 = acc0;
  for (int kb = 0; kb < Kd; kb += 32) {
    // A layout (matches B): elems 0..15 -> K=kb+16*hi+j
    v8h alo = *(const v8h*)(ap + kb + 16 * hi);
    v8h ahi = *(const v8h*)(ap + kb + 16 * hi + 8);
    v16h a;
#pragma unroll
    for (int j = 0; j < 8; ++j) { a[j] = alo[j]; a[8 + j] = ahi[j]; }
    // B layout: lanes 0-15 K=kb..kb+15, lanes 16-31 K=kb+16..kb+31 (32B contiguous per lane)
    v16h b0, b1, b2, b3;
    v8h t0 = *(const v8h*)(w0 + kb + 16 * hi), t1 = *(const v8h*)(w0 + kb + 16 * hi + 8);
    v8h u0 = *(const v8h*)(w1 + kb + 16 * hi), u1 = *(const v8h*)(w1 + kb + 16 * hi + 8);
    v8h s0 = *(const v8h*)(w2 + kb + 16 * hi), s1 = *(const v8h*)(w2 + kb + 16 * hi + 8);
    v8h r0 = *(const v8h*)(w3 + kb + 16 * hi), r1 = *(const v8h*)(w3 + kb + 16 * hi + 8);
#pragma unroll
    for (int j = 0; j < 8; ++j) {
      b0[j] = t0[j]; b0[8 + j] = t1[j];
      b1[j] = u0[j]; b1[8 + j] = u1[j];
      b2[j] = s0[j]; b2[8 + j] = s1[j];
      b3[j] = r0[j]; b3[8 + j] = r1[j];
    }
    acc0 = __builtin_amdgcn_wmma_f32_16x16x32_f16(false, a, false, b0, (short)0, acc0, false, false);
    acc1 = __builtin_amdgcn_wmma_f32_16x16x32_f16(false, a, false, b1, (short)0, acc1, false, false);
    acc2 = __builtin_amdgcn_wmma_f32_16x16x32_f16(false, a, false, b2, (short)0, acc2, false, false);
    acc3 = __builtin_amdgcn_wmma_f32_16x16x32_f16(false, a, false, b3, (short)0, acc3, false, false);
  }
#pragma unroll
  for (int t = 0; t < 4; ++t) {
    v8f acc = (t == 0) ? acc0 : (t == 1) ? acc1 : (t == 2) ? acc2 : acc3;
    int col = bcol0 + 16 * t;
#pragma unroll
    for (int r = 0; r < 8; ++r) {
      int orow = blockIdx.y * 16 + hi * 8 + r;   // C/D layout: VGPR r -> row 8*hi+r
      if (orow < n) {
        float y = acc[r] + bias[col];
        float e = (y > 0.0f) ? y : expm1f(y);    // ELU (alpha=1)
        float o;
        if (mode == 0)
          o = bng[col] * (e - bnm[col]) * rsqrtf(bnv[col] + 1e-5f) + bnb[col];
        else
          o = e;
        H[(size_t)orow * co + col] = o;
      }
    }
  }
}

// ---------- voxel pooling kernels ----------
__global__ void cluster_kernel(const float* __restrict__ pos, const int* __restrict__ batch,
                               int batchdiv, float vsize, int nc, int* __restrict__ cluster,
                               int n) {
  int i = blockIdx.x * blockDim.x + threadIdx.x;
  if (i >= n) return;
  int b = batch ? batch[i] : (i / batchdiv);
  int cc[3];
#pragma unroll
  for (int d = 0; d < 3; ++d) {
    int v = (int)floorf(pos[(size_t)i * 3 + d] / vsize);
    v = v < 0 ? 0 : (v > nc - 1 ? nc - 1 : v);
    cc[d] = v;
  }
  cluster[i] = b * nc * nc * nc + cc[0] + nc * cc[1] + nc * nc * cc[2];
}

__global__ void poolmax_kernel(const float* __restrict__ H, const int* __restrict__ cluster,
                               const int* __restrict__ valid, unsigned* __restrict__ pxbits,
                               int c, size_t total) {
  size_t idx = (size_t)blockIdx.x * blockDim.x + threadIdx.x;
  if (idx >= total) return;
  int node = (int)(idx / c);
  int ch = (int)(idx % c);
  if (valid && !valid[node]) return;
  atomicMax(&pxbits[(size_t)cluster[node] * c + ch], fenc(H[idx]));
}

__global__ void poolcnt_kernel(const int* __restrict__ cluster, const int* __restrict__ valid,
                               const float* __restrict__ pos, float* __restrict__ cnt,
                               float* __restrict__ possum, int n) {
  int i = blockIdx.x * blockDim.x + threadIdx.x;
  if (i >= n) return;
  if (valid && !valid[i]) return;
  int cl = cluster[i];
  atomicAdd(&cnt[cl], 1.0f);
#pragma unroll
  for (int d = 0; d < 3; ++d)
    atomicAdd(&possum[(size_t)cl * 3 + d], pos[(size_t)i * 3 + d]);
}

__global__ void finfeat_kernel(const unsigned* __restrict__ pxbits, const float* __restrict__ cnt,
                               float* __restrict__ px, int c, size_t total) {
  size_t idx = (size_t)blockIdx.x * blockDim.x + threadIdx.x;
  if (idx >= total) return;
  size_t j = idx / c;
  px[idx] = (cnt[j] > 0.0f) ? fdec(pxbits[idx]) : 0.0f;
}

__global__ void finpos_kernel(const float* __restrict__ possum, const float* __restrict__ cnt,
                              float* __restrict__ ppos, int* __restrict__ nvalid, int M) {
  int j = blockIdx.x * blockDim.x + threadIdx.x;
  if (j >= M) return;
  float cv = cnt[j];
  nvalid[j] = (cv > 0.0f) ? 1 : 0;
  float ic = 1.0f / fmaxf(cv, 1.0f);
#pragma unroll
  for (int d = 0; d < 3; ++d)
    ppos[(size_t)j * 3 + d] = possum[(size_t)j * 3 + d] * ic;
}

// Remap edges to cluster ids, drop merged self loops, compute Cartesian edge attr
// and the global max|cart| (non-negative -> raw float bits are uint-monotone).
__global__ void remap_kernel(const int* __restrict__ src, const int* __restrict__ dst,
                             const float* __restrict__ emask, const int* __restrict__ cluster,
                             const float* __restrict__ ppos, int* __restrict__ nsrc,
                             int* __restrict__ ndst, float* __restrict__ nemask,
                             float* __restrict__ cart, unsigned* __restrict__ maxabs, int ne) {
  int e = blockIdx.x * blockDim.x + threadIdx.x;
  if (e >= ne) return;
  int ns = cluster[src[e]];
  int nd = cluster[dst[e]];
  float em = emask ? emask[e] : 1.0f;
  float nem = (ns != nd) ? em : 0.0f;
  nsrc[e] = ns; ndst[e] = nd; nemask[e] = nem;
  float mx = 0.0f;
#pragma unroll
  for (int d = 0; d < 3; ++d) {
    float cv = (ppos[(size_t)nd * 3 + d] - ppos[(size_t)ns * 3 + d]) * nem;
    cart[(size_t)e * 3 + d] = cv;
    mx = fmaxf(mx, fabsf(cv));
  }
  atomicMax(maxabs, __float_as_uint(mx));
}

__global__ void u_kernel(const float* __restrict__ cart, const unsigned* __restrict__ maxabs,
                         float* __restrict__ u, int total) {
  int i = blockIdx.x * blockDim.x + threadIdx.x;
  if (i >= total) return;
  float m = fmaxf(__uint_as_float(*maxabs), 1e-12f);
  u[i] = cart[i] / (2.0f * m) + 0.5f;
}

// Stable "non-empty clusters first" selection of 64 voxels per graph, copy features.
__global__ void select_kernel(const float* __restrict__ px, const float* __restrict__ cnt,
                              float* __restrict__ xf) {
  __shared__ int sel[64];
  int b = blockIdx.x;
  if (threadIdx.x == 0) {
    int c = 0;
    for (int j = 0; j < 125 && c < 64; ++j)
      if (cnt[b * 125 + j] > 0.0f) sel[c++] = j;
    for (int j = 0; j < 125 && c < 64; ++j)
      if (!(cnt[b * 125 + j] > 0.0f)) sel[c++] = j;
  }
  __syncthreads();
  for (int t = threadIdx.x; t < 64 * 512; t += blockDim.x) {
    int s = t >> 9, ch = t & 511;
    int cl = b * 125 + sel[s];
    xf[(size_t)b * 32768 + t] = (cnt[cl] > 0.0f) ? px[(size_t)cl * 512 + ch] : 0.0f;
  }
}

__global__ void fc2_kernel(const float* __restrict__ h1, const float* __restrict__ fW2,
                           const float* __restrict__ fb2, float* __restrict__ out) {
  int t = threadIdx.x;
  if (t >= 8) return;
  int b = t >> 1, o = t & 1;
  float s = fb2[o];
  for (int k = 0; k < 1024; ++k) s += h1[(size_t)b * 1024 + k] * fW2[(size_t)k * 2 + o];
  out[b * 2 + o] = s;
}

// ---------- host orchestration ----------
extern "C" void kernel_launch(void* const* d_in, const int* in_sizes, int n_in,
                              void* d_out, int out_size, void* d_ws, size_t ws_size,
                              hipStream_t stream) {
  (void)in_sizes; (void)n_in; (void)out_size; (void)ws_size;

  const float* x0     = (const float*)d_in[0];
  const float* pos0   = (const float*)d_in[1];
  const int*   batch0 = (const int*)d_in[2];
  const int*   ei     = (const int*)d_in[3];
  const int*   src0   = ei;
  const int*   dst0   = ei + NE;
  const float* u0     = (const float*)d_in[4];
  const float* fW1    = (const float*)d_in[33];
  const float* fb1    = (const float*)d_in[34];
  const float* fW2    = (const float*)d_in[35];
  const float* fb2    = (const float*)d_in[36];

  // --- workspace arena ---
  char* base = (char*)d_ws;
  size_t off = 0;
  auto arena = [&](size_t bytes) -> char* {
    off = (off + 255) & ~(size_t)255;
    char* p = base + off;
    off += bytes;
    return p;
  };
  float*     S    = (float*)arena((size_t)37044 * 1728 * 4);      // max n*27*ci (layer 2)
  _Float16*  Ap   = (_Float16*)arena((size_t)37056 * 1792 * 2);   // max n16*Kd (layer 2)
  _Float16*  Wp   = (_Float16*)arena((size_t)1024 * 32768 * 2);   // max co*Kd (FC1)
  float*     H    = (float*)arena((size_t)37056 * 128 * 4);       // max n16*co (layer 2)
  float*     PX   = (float*)arena((size_t)37044 * 64 * 4);        // max M*c (pool 1)
  unsigned*  PXB  = (unsigned*)arena((size_t)37044 * 64 * 4);
  float*   possum = (float*)arena((size_t)37044 * 3 * 4);
  float*     posA = (float*)arena((size_t)37044 * 3 * 4);
  float*     posB = (float*)arena((size_t)37044 * 3 * 4);
  float*     cnt  = (float*)arena((size_t)37044 * 4);
  float*     deg  = (float*)arena((size_t)37044 * 4);
  int*     clus   = (int*)arena((size_t)37044 * 4);
  int*       nvA  = (int*)arena((size_t)37044 * 4);
  int*       nvB  = (int*)arena((size_t)37044 * 4);
  float*   basis  = (float*)arena((size_t)NE * 8 * 4);
  int*      kidx  = (int*)arena((size_t)NE * 8 * 4);
  int*       esA  = (int*)arena((size_t)NE * 4);
  int*       edA  = (int*)arena((size_t)NE * 4);
  int*       esB  = (int*)arena((size_t)NE * 4);
  int*       edB  = (int*)arena((size_t)NE * 4);
  float*     emA  = (float*)arena((size_t)NE * 4);
  float*     emB  = (float*)arena((size_t)NE * 4);
  float*    cart  = (float*)arena((size_t)NE * 3 * 4);
  float*      uA  = (float*)arena((size_t)NE * 3 * 4);
  float*      uB  = (float*)arena((size_t)NE * 3 * 4);
  unsigned* maxab = (unsigned*)arena(256);
  float*      xf  = (float*)arena((size_t)BG * 32768 * 4);
  float*      h1  = (float*)arena((size_t)16 * 1024 * 4);

  auto nb = [](size_t total, int bs) { return (unsigned)((total + bs - 1) / bs); };

  // spline-conv layer: scatter -> pack A/W to f16 -> WMMA GEMM (fused ELU+BN)
  auto launch_conv = [&](const float* Xin, const int* src, const int* dst,
                         const float* uptr, const float* em, int n, int ci, int co, int li) {
    int k1 = 27 * ci;
    int Kd = ((k1 + ci + 31) / 32) * 32;
    int n16 = (n + 15) & ~15;
    const float* W  = (const float*)d_in[5 + 7 * li + 0];
    const float* Wr = (const float*)d_in[5 + 7 * li + 1];
    const float* cb = (const float*)d_in[5 + 7 * li + 2];
    const float* g  = (const float*)d_in[5 + 7 * li + 3];
    const float* be = (const float*)d_in[5 + 7 * li + 4];
    const float* m  = (const float*)d_in[5 + 7 * li + 5];
    const float* v  = (const float*)d_in[5 + 7 * li + 6];
    hipMemsetAsync(deg, 0, (size_t)n * 4, stream);
    hipMemsetAsync(S, 0, (size_t)n * 27 * ci * 4, stream);
    basis_kernel<<<nb(NE, 256), 256, 0, stream>>>(uptr, em, dst, basis, kidx, deg, NE);
    size_t st = (size_t)NE * ci;
    scatter_kernel<<<nb(st, 256), 256, 0, stream>>>(Xin, src, dst, basis, kidx, S, ci, st);
    size_t at = (size_t)n16 * Kd;
    pack_a_kernel<<<nb(at, 256), 256, 0, stream>>>(S, Xin, deg, Ap, n, k1, ci, Kd, at);
    size_t wt = (size_t)co * Kd;
    pack_w_kernel<<<nb(wt, 256), 256, 0, stream>>>(W, Wr, Wp, k1, ci, Kd, co);
    dim3 grid(co / 64, n16 / 16);
    wmma_gemm_kernel<<<grid, 32, 0, stream>>>(Ap, Wp, cb, g, be, m, v, H, n, Kd, co, 0);
  };

  // voxel pool: cluster -> segmented max/sum -> finalize -> edge remap + Cartesian u
  auto launch_pool = [&](int n, int c, float vsize, int ncg, const float* posIn,
                         const int* batchArr, int batchdiv, const int* validIn,
                         float* pposOut, int* nvOut, const int* srcIn, const int* dstIn,
                         const float* emIn, int* srcOut, int* dstOut, float* emOut,
                         float* uOut, bool doEdges) {
    int M = BG * ncg * ncg * ncg;
    cluster_kernel<<<nb(n, 256), 256, 0, stream>>>(posIn, batchArr, batchdiv, vsize, ncg,
                                                   clus, n);
    size_t mt = (size_t)M * c;
    unsigned initv;
    { float f = -1e30f; unsigned ub; memcpy(&ub, &f, 4); initv = ~ub; }   // fenc(NEG)
    fill_u32_kernel<<<nb(mt, 256), 256, 0, stream>>>(PXB, initv, mt);
    hipMemsetAsync(cnt, 0, (size_t)M * 4, stream);
    hipMemsetAsync(possum, 0, (size_t)M * 3 * 4, stream);
    size_t nt = (size_t)n * c;
    poolmax_kernel<<<nb(nt, 256), 256, 0, stream>>>(H, clus, validIn, PXB, c, nt);
    poolcnt_kernel<<<nb(n, 256), 256, 0, stream>>>(clus, validIn, posIn, cnt, possum, n);
    finfeat_kernel<<<nb(mt, 256), 256, 0, stream>>>(PXB, cnt, PX, c, mt);
    finpos_kernel<<<nb(M, 256), 256, 0, stream>>>(possum, cnt, pposOut, nvOut, M);
    if (doEdges) {
      hipMemsetAsync(maxab, 0, 4, stream);
      remap_kernel<<<nb(NE, 256), 256, 0, stream>>>(srcIn, dstIn, emIn, clus, pposOut,
                                                    srcOut, dstOut, emOut, cart, maxab, NE);
      u_kernel<<<nb((size_t)NE * 3, 256), 256, 0, stream>>>(cart, maxab, uOut, NE * 3);
    }
  };

  // ---- layer 1 (n=16384, 4->64) + pool(0.05, nc=21, M=37044) ----
  launch_conv(x0, src0, dst0, u0, nullptr, NPT, 4, 64, 0);
  launch_pool(NPT, 64, 0.05f, 21, pos0, batch0, 1, nullptr, posA, nvA,
              src0, dst0, nullptr, esA, edA, emA, uA, true);
  // ---- layer 2 (n=37044, 64->128) + pool(0.1, nc=11, M=5324) ----
  launch_conv(PX, esA, edA, uA, emA, 37044, 64, 128, 1);
  launch_pool(37044, 128, 0.1f, 11, posA, nullptr, 9261, nvA, posB, nvB,
              esA, edA, emA, esB, edB, emB, uB, true);
  // ---- layer 3 (n=5324, 128->256) + pool(0.15, nc=7, M=1372) ----
  launch_conv(PX, esB, edB, uB, emB, 5324, 128, 256, 2);
  launch_pool(5324, 256, 0.15f, 7, posB, nullptr, 1331, nvB, posA, nvA,
              esB, edB, emB, esA, edA, emA, uA, true);
  // ---- layer 4 (n=1372, 256->512) + final pool(0.25, nc=5, M=500, feats only) ----
  launch_conv(PX, esA, edA, uA, emA, 1372, 256, 512, 3);
  launch_pool(1372, 512, 0.25f, 5, posA, nullptr, 343, nvA, posB, nvB,
              nullptr, nullptr, nullptr, nullptr, nullptr, nullptr, nullptr, false);

  // ---- select 64 voxels/graph, FC1 via WMMA, FC2 scalar ----
  select_kernel<<<BG, 256, 0, stream>>>(PX, cnt, xf);
  {
    // A = xf (4 real rows padded to 16), K = 32768, co = 1024
    size_t at = (size_t)16 * 32768;
    pack_a_kernel<<<nb(at, 256), 256, 0, stream>>>(xf, xf, nullptr, Ap, 4, 0, 32768, 32768, at);
    size_t wt = (size_t)1024 * 32768;
    pack_w_kernel<<<nb(wt, 256), 256, 0, stream>>>(fW1, fW1, Wp, 0, 32768, 32768, 1024);
    dim3 grid(1024 / 64, 1);
    wmma_gemm_kernel<<<grid, 32, 0, stream>>>(Ap, Wp, fb1, fb1, fb1, fb1, fb1, h1,
                                              4, 32768, 1024, 1);
  }
  fc2_kernel<<<1, 8, 0, stream>>>(h1, fW2, fb2, (float*)d_out);
}
